// ResidueGAT_48644799595236
// MI455X (gfx1250) — matrix operand order repaired and checked by reading, not compiled, not run on hardware
//
#include <hip/hip_runtime.h>

// ---------------------------------------------------------------------------
// ResidueGAT for MI455X (gfx1250, wave32, WMMA).
//   GEMM xl = x@W        : v_wmma_f32_16x16x32_f16, f32 accum
//   attention softmax    : atomic segment-max + fused exp/scatter-add (L2-resident)
//   MLP                  : v_wmma_f32_16x16x32_f16 + fused bias/ReLU epilogue
// ---------------------------------------------------------------------------

#define N_NODES 50000
#define N_EDGES 800000
#define IN_F    256
#define HEADS   4
#define CH      128
#define HC      512   // HEADS*CH
#define NEG_SLOPE 0.2f

typedef _Float16 half8 __attribute__((ext_vector_type(8)));
typedef _Float16 v16h  __attribute__((ext_vector_type(16)));
typedef float    v8f   __attribute__((ext_vector_type(8)));
typedef float    f4    __attribute__((ext_vector_type(4)));

// ---------------- WMMA fragment loaders (ISA 7.12.2 layouts) ----------------
// A: 16x32 f16, row M = mrow + (lane&15).
//   lane<16 : halves 0-7 = K 0..7,  halves 8-15 = K 16..23
//   lane>=16: halves 0-7 = K 8..15, halves 8-15 = K 24..31
__device__ inline v16h load_a_frag(const _Float16* A, int lda, int mrow, int kb, int lane) {
  int m  = mrow + (lane & 15);
  int k0 = kb + ((lane >> 4) << 3);
  const _Float16* p = A + (size_t)m * lda + k0;
  half8 lo = *(const half8*)(p);
  half8 hi = *(const half8*)(p + 16);
  return __builtin_shufflevector(lo, hi, 0,1,2,3,4,5,6,7,8,9,10,11,12,13,14,15);
}

// B: 32x16 f16, column N = ncol + (lane&15), loaded from transposed (N-major)
// weights so K values per lane are contiguous in 8-element runs.
//   lane<16 : K kb..kb+7, kb+16..kb+23 ; lane>=16 : K kb+8..kb+15, kb+24..kb+31
__device__ inline v16h load_b_frag(const _Float16* BT, int ldb, int ncol, int kb, int lane) {
  int n  = ncol + (lane & 15);
  int k0 = kb + ((lane >> 4) << 3);
  const _Float16* p = BT + (size_t)n * ldb + k0;
  half8 lo = *(const half8*)(p);
  half8 hi = *(const half8*)(p + 16);
  return __builtin_shufflevector(lo, hi, 0,1,2,3,4,5,6,7,8,9,10,11,12,13,14,15);
}

// ---------------- conversion / init kernels ----------------
__global__ void k_cvt_x(const float* __restrict__ x, _Float16* __restrict__ xh, int n) {
  int i = blockIdx.x * blockDim.x + threadIdx.x;
  if (i < n) xh[i] = (_Float16)x[i];
}

// W [256,512] (k-major rows) -> WT [512,256] (n-major rows, K contiguous)
__global__ void k_cvt_wT(const float* __restrict__ W, _Float16* __restrict__ WT) {
  int i = blockIdx.x * blockDim.x + threadIdx.x;
  if (i < HC * IN_F) {
    int n = i / IN_F, k = i % IN_F;
    WT[i] = (_Float16)W[(size_t)k * HC + n];
  }
}

// w1 [128,128] (in,out) -> w1T [128,128] (out-major, K=in contiguous)
__global__ void k_cvt_w1T(const float* __restrict__ w1, _Float16* __restrict__ w1T) {
  int i = blockIdx.x * blockDim.x + threadIdx.x;
  if (i < CH * CH) {
    int o = i / CH, k = i % CH;
    w1T[i] = (_Float16)w1[(size_t)k * CH + o];
  }
}

__global__ void k_zero(float* __restrict__ agg, float* __restrict__ mmax,
                       float* __restrict__ denom) {
  int i = blockIdx.x * blockDim.x + threadIdx.x;
  if (i < N_NODES * HC) agg[i] = 0.0f;
  if (i < N_NODES * HEADS) { mmax[i] = -INFINITY; denom[i] = 0.0f; }
}

// ---------------- GEMM 1: xl = x @ W  (f16 WMMA, f32 out) ----------------
// block = 256 threads = 8 waves; wave computes 16x64; block covers 16 rows x 512 cols.
__global__ __launch_bounds__(256) void k_gemm_xw(const _Float16* __restrict__ xh,
                                                 const _Float16* __restrict__ WT,
                                                 float* __restrict__ xl) {
  int lane = threadIdx.x & 31;
  int wave = threadIdx.x >> 5;
  int mrow = blockIdx.x * 16;      // 50000/16 = 3125 blocks, exact
  int ncol = wave * 64;
  v8f acc[4] = {};
  for (int kb = 0; kb < IN_F; kb += 32) {
    if (kb + 32 < IN_F)  // gfx1250 global_prefetch_b8 of next A tile row
      __builtin_prefetch(xh + (size_t)(mrow + (lane & 15)) * IN_F + kb + 32, 0, 1);
    v16h a = load_a_frag(xh, IN_F, mrow, kb, lane);
#pragma unroll
    for (int s = 0; s < 4; ++s) {
      v16h b = load_b_frag(WT, IN_F, ncol + 16 * s, kb, lane);
      acc[s] = __builtin_amdgcn_wmma_f32_16x16x32_f16(
          false, a, false, b, (short)0, acc[s], false, false);
    }
  }
  // C layout: VGPR r -> row M = r (lanes 0-15) or 8+r (lanes 16-31), N = lane&15
  int rbase = mrow + ((lane >> 4) << 3);
  int nc = lane & 15;
#pragma unroll
  for (int s = 0; s < 4; ++s) {
    int col = ncol + 16 * s + nc;
#pragma unroll
    for (int r = 0; r < 8; ++r)
      xl[(size_t)(rbase + r) * HC + col] = acc[s][r];
  }
}

// ---------------- attention coefficients: a_src/a_dst [N,H] ----------------
__global__ __launch_bounds__(256) void k_att(const float* __restrict__ xl,
                                             const float* __restrict__ att_src,
                                             const float* __restrict__ att_dst,
                                             float* __restrict__ a_src,
                                             float* __restrict__ a_dst) {
  int lane = threadIdx.x & 31;
  int w = blockIdx.x * 8 + (threadIdx.x >> 5);
  if (w >= N_NODES * HEADS) return;
  int n = w >> 2, h = w & 3;
  const float* xp = xl + (size_t)n * HC + h * CH;
  const float* as = att_src + h * CH;
  const float* ad = att_dst + h * CH;
  float ss = 0.f, sd = 0.f;
#pragma unroll
  for (int j = 0; j < 4; ++j) {
    int c = lane + 32 * j;
    float v = xp[c];
    ss += v * as[c];
    sd += v * ad[c];
  }
#pragma unroll
  for (int off = 16; off > 0; off >>= 1) {
    ss += __shfl_xor(ss, off, 32);
    sd += __shfl_xor(sd, off, 32);
  }
  if (lane == 0) { a_src[w] = ss; a_dst[w] = sd; }
}

// ---------------- segment max over dst (float atomic-max bit trick) --------
__device__ inline void atomicMaxF(float* addr, float v) {
  if (v >= 0.f) atomicMax((int*)addr, __float_as_int(v));
  else          atomicMin((unsigned int*)addr, __float_as_uint(v));
}

__global__ void k_edge_max(const long long* __restrict__ ei,
                           const float* __restrict__ a_src,
                           const float* __restrict__ a_dst,
                           float* __restrict__ mmax) {
  int idx = blockIdx.x * blockDim.x + threadIdx.x;
  const int tot = (N_EDGES + N_NODES) * HEADS;
  if (idx >= tot) return;
  int e = idx >> 2, h = idx & 3;
  int s, d;
  if (e < N_EDGES) { s = (int)ei[e]; d = (int)ei[N_EDGES + e]; }
  else             { s = d = e - N_EDGES; }  // self loop
  float l = a_src[s * 4 + h] + a_dst[d * 4 + h];
  l = l > 0.f ? l : NEG_SLOPE * l;
  atomicMaxF(&mmax[d * 4 + h], l);
}

// ---------------- fused exp + weighted scatter-add (wave per edge) ---------
__global__ __launch_bounds__(256) void k_edge_scatter(const long long* __restrict__ ei,
                                                      const float* __restrict__ a_src,
                                                      const float* __restrict__ a_dst,
                                                      const float* __restrict__ mmax,
                                                      const float* __restrict__ xl,
                                                      float* __restrict__ denom,
                                                      float* __restrict__ agg) {
  int lane = threadIdx.x & 31;
  int e = blockIdx.x * 8 + (threadIdx.x >> 5);
  if (e >= N_EDGES + N_NODES) return;
  int s, d;
  if (e < N_EDGES) { s = (int)ei[e]; d = (int)ei[N_EDGES + e]; }
  else             { s = d = e - N_EDGES; }
  const float* xs = xl + (size_t)s * HC;
  float* ag = agg + (size_t)d * HC;
#pragma unroll
  for (int h = 0; h < HEADS; ++h) {
    float l = a_src[s * 4 + h] + a_dst[d * 4 + h];
    l = l > 0.f ? l : NEG_SLOPE * l;
    float wgt = expf(l - mmax[d * 4 + h]);
    if (lane == 0) atomicAdd(&denom[d * 4 + h], wgt);
    f4 xv = *(const f4*)(xs + h * CH + lane * 4);   // coalesced 512B/wave
    float* ap = ag + h * CH + lane * 4;
    atomicAdd(ap + 0, wgt * xv[0]);
    atomicAdd(ap + 1, wgt * xv[1]);
    atomicAdd(ap + 2, wgt * xv[2]);
    atomicAdd(ap + 3, wgt * xv[3]);
  }
}

// ---------------- head mean + bias + ReLU -> f16 h [N,128] -----------------
__global__ void k_final_gat(const float* __restrict__ agg,
                            const float* __restrict__ denom,
                            const float* __restrict__ gat_bias,
                            _Float16* __restrict__ hf) {
  int idx = blockIdx.x * blockDim.x + threadIdx.x;
  if (idx >= N_NODES * CH) return;
  int n = idx >> 7, c = idx & 127;
  float acc = 0.f;
#pragma unroll
  for (int h = 0; h < HEADS; ++h)
    acc += agg[(size_t)n * HC + h * CH + c] / denom[n * 4 + h];
  float v = 0.25f * acc + gat_bias[c];
  v = v > 0.f ? v : 0.f;
  hf[idx] = (_Float16)v;
}

// ---------------- GEMM 2: h2 = relu(h @ w1 + b1)  (f16 WMMA) ---------------
// wave = 16x64; 2 waves per 16-row tile; block of 8 waves = 4 tiles.
__global__ __launch_bounds__(256) void k_gemm_mlp1(const _Float16* __restrict__ hf,
                                                   const _Float16* __restrict__ w1T,
                                                   const float* __restrict__ b1,
                                                   _Float16* __restrict__ h2f) {
  int lane = threadIdx.x & 31;
  int wave = threadIdx.x >> 5;
  int mrow = (blockIdx.x * 4 + (wave >> 1)) * 16;
  if (mrow >= N_NODES) return;               // wave-uniform guard (EXEC stays all-1s)
  int ncol = (wave & 1) * 64;
  v8f acc[4] = {};
  for (int kb = 0; kb < CH; kb += 32) {
    v16h a = load_a_frag(hf, CH, mrow, kb, lane);
#pragma unroll
    for (int s = 0; s < 4; ++s) {
      v16h b = load_b_frag(w1T, CH, ncol + 16 * s, kb, lane);
      acc[s] = __builtin_amdgcn_wmma_f32_16x16x32_f16(
          false, a, false, b, (short)0, acc[s], false, false);
    }
  }
  int rbase = mrow + ((lane >> 4) << 3);
  int nc = lane & 15;
#pragma unroll
  for (int s = 0; s < 4; ++s) {
    int col = ncol + 16 * s + nc;
    float bias = b1[col];
#pragma unroll
    for (int r = 0; r < 8; ++r) {
      float v = acc[s][r] + bias;
      v = v > 0.f ? v : 0.f;
      h2f[(size_t)(rbase + r) * CH + col] = (_Float16)v;
    }
  }
}

// ---------------- out[n] = h2[n,:] . w2 + b2 (wave per node) ---------------
__global__ __launch_bounds__(256) void k_out(const _Float16* __restrict__ h2f,
                                             const float* __restrict__ w2,
                                             const float* __restrict__ b2,
                                             float* __restrict__ out) {
  int lane = threadIdx.x & 31;
  int n = blockIdx.x * 8 + (threadIdx.x >> 5);
  if (n >= N_NODES) return;
  float s = 0.f;
#pragma unroll
  for (int j = 0; j < 4; ++j) {
    int c = lane + 32 * j;
    s += (float)h2f[(size_t)n * CH + c] * w2[c];
  }
#pragma unroll
  for (int off = 16; off > 0; off >>= 1) s += __shfl_xor(s, off, 32);
  if (lane == 0) out[n] = s + b2[0];
}

// ---------------------------------------------------------------------------
extern "C" void kernel_launch(void* const* d_in, const int* in_sizes, int n_in,
                              void* d_out, int out_size, void* d_ws, size_t ws_size,
                              hipStream_t stream) {
  (void)in_sizes; (void)n_in; (void)out_size; (void)ws_size;
  const float*     x        = (const float*)d_in[0];
  const long long* ei       = (const long long*)d_in[1];
  const float*     W        = (const float*)d_in[2];
  const float*     att_src  = (const float*)d_in[3];
  const float*     att_dst  = (const float*)d_in[4];
  const float*     gat_bias = (const float*)d_in[5];
  const float*     w1       = (const float*)d_in[6];
  const float*     b1       = (const float*)d_in[7];
  const float*     w2       = (const float*)d_in[8];
  const float*     b2       = (const float*)d_in[9];
  float* out = (float*)d_out;

  char* base = (char*)d_ws;
  size_t off = 0;
  auto take = [&](size_t bytes) -> void* {
    void* p = base + off;
    off += (bytes + 255) & ~(size_t)255;
    return p;
  };
  _Float16* xh   = (_Float16*)take((size_t)N_NODES * IN_F * 2);   // 25.6 MB
  _Float16* WT   = (_Float16*)take((size_t)HC * IN_F * 2);        // 256 KB
  _Float16* w1T  = (_Float16*)take((size_t)CH * CH * 2);          // 32 KB
  float*    xl   = (float*)take((size_t)N_NODES * HC * 4);        // 102.4 MB (L2)
  float*    aS   = (float*)take((size_t)N_NODES * HEADS * 4);
  float*    aD   = (float*)take((size_t)N_NODES * HEADS * 4);
  float*    mmax = (float*)take((size_t)N_NODES * HEADS * 4);
  float*    den  = (float*)take((size_t)N_NODES * HEADS * 4);
  float*    agg  = (float*)take((size_t)N_NODES * HC * 4);        // 102.4 MB (L2)
  _Float16* hf   = (_Float16*)take((size_t)N_NODES * CH * 2);
  _Float16* h2f  = (_Float16*)take((size_t)N_NODES * CH * 2);

  const int T = 256;
  // precision conversions / transposes
  k_cvt_x  <<<(N_NODES * IN_F + T - 1) / T, T, 0, stream>>>(x, xh, N_NODES * IN_F);
  k_cvt_wT <<<(HC * IN_F + T - 1) / T, T, 0, stream>>>(W, WT);
  k_cvt_w1T<<<(CH * CH + T - 1) / T, T, 0, stream>>>(w1, w1T);
  // zero accumulators (required every replay: graph re-executes this launch)
  k_zero   <<<(N_NODES * HC + T - 1) / T, T, 0, stream>>>(agg, mmax, den);
  // xl = x @ W (WMMA)
  k_gemm_xw<<<N_NODES / 16, T, 0, stream>>>(xh, WT, xl);
  // attention dots
  k_att    <<<(N_NODES * HEADS + 7) / 8, T, 0, stream>>>(xl, att_src, att_dst, aS, aD);
  // segment softmax (max pass, then fused exp + scatter)
  k_edge_max    <<<((N_EDGES + N_NODES) * HEADS + T - 1) / T, T, 0, stream>>>(ei, aS, aD, mmax);
  k_edge_scatter<<<(N_EDGES + N_NODES + 7) / 8, T, 0, stream>>>(ei, aS, aD, mmax, xl, den, agg);
  // head mean + bias + relu
  k_final_gat<<<(N_NODES * CH + T - 1) / T, T, 0, stream>>>(agg, den, gat_bias, hf);
  // MLP layer 1 (WMMA) + final dot
  k_gemm_mlp1<<<(N_NODES / 16 + 3) / 4, T, 0, stream>>>(hf, w1T, b1, h2f);
  k_out      <<<(N_NODES + 7) / 8, T, 0, stream>>>(h2f, w2, b2, out);
}
